// LyricSTM_54305566491412
// MI455X (gfx1250) — matrix operand order repaired and checked by reading, not compiled, not run on hardware
//
#include <hip/hip_runtime.h>
#include <math.h>

typedef __attribute__((ext_vector_type(16))) __bf16 v16bf;
typedef __attribute__((ext_vector_type(8)))  float  v8f;
typedef unsigned short u16;

#define T_STEPS 512
#define BATCH   64
#define FDIM    512
#define HDIM    1024
#define MLPDIM  128
#define KTOT    (FDIM + HDIM)       // 1536
#define APAD    (KTOT + 8)          // 1544: +16B row pad -> conflict-free LDS frags
#define HPAD    (HDIM + 8)          // 1032
#define H1PAD   (MLPDIM + 8)        // 136

// ---------- helpers ----------

static __device__ __forceinline__ u16 f32_to_bf16(float f) {
    unsigned int u = __float_as_uint(f);
    u += 0x7FFFu + ((u >> 16) & 1u);   // round-to-nearest-even
    return (u16)(u >> 16);
}

static __device__ __forceinline__ float sigmoidf_(float x) {
    return 1.0f / (1.0f + __expf(-x));
}

// A fragment, 16x32 bf16 (M x K): lane holds row m=L%16;
// lanes 0-15 K = {kb+0..7, kb+16..23}, lanes 16-31 +8 (kl = (L>=16)*8).
static __device__ __forceinline__ v16bf frag_a(const u16* row, int k0, int kl) {
    union { uint4 q[2]; v16bf v; } f;
    f.q[0] = *(const uint4*)(row + k0 + kl);
    f.q[1] = *(const uint4*)(row + k0 + kl + 16);
    return f.v;
}

// B fragment, 32x16 bf16 (K x N): lane holds col n=L%16;
// lanes 0-15 K = {kb+0..7, kb+16..23}, lanes 16-31 +8 (ko = (L>=16)*16, offset ko/2).
static __device__ __forceinline__ v16bf frag_b(const u16* row, int k0, int ko) {
    union { uint4 q[2]; v16bf v; } f;
    f.q[0] = *(const uint4*)(row + k0 + (ko >> 1));
    f.q[1] = *(const uint4*)(row + k0 + (ko >> 1) + 16);
    return f.v;
}

static __device__ __forceinline__ v8f wmma_bf16(v16bf a, v16bf b, v8f c) {
    return __builtin_amdgcn_wmma_f32_16x16x32_bf16(false, a, false, b, (short)0, c, false, false);
}

// ---------- one-time prep kernels ----------

__global__ void k_cvt_bf16(const float* __restrict__ src, u16* __restrict__ dst, int n) {
    int i = blockIdx.x * blockDim.x + threadIdx.x;
    int stride = gridDim.x * blockDim.x;
    for (; i < n; i += stride) dst[i] = f32_to_bf16(src[i]);
}

__global__ void k_bias_and_zero(const float* __restrict__ bih, const float* __restrict__ bhh,
                                float* __restrict__ bsum, float* __restrict__ C) {
    int i = blockIdx.x * blockDim.x + threadIdx.x;   // 65536 threads
    if (i < 4 * HDIM) bsum[i] = bih[i] + bhh[i];
    C[i] = 0.0f;
}

// ---------- fused LSTM step: gates GEMM + cell update ----------
// grid = 64 WGs (16 hidden cols each), block = 256 (8 waves).
// A = [x_t | h_{t-1}] (64x1536 bf16) staged once in LDS (192 KB, CDNA5 320KB/WGP).
// wave w: gate g = w&3, K-half kh = w>>2, all 4 M-tiles.
// -> each weight row element loaded from L2 exactly once per WG (full B dedup),
//    A fragments served from LDS (ds_load_b128), partial sums reduced via LDS.
__global__ __launch_bounds__(256) void k_lstm_step(
    const u16* __restrict__ Xbf,        // [T,B,F] bf16
    u16* __restrict__ Hs,               // [T,B,H] bf16 (h history)
    const u16* __restrict__ Wih,        // [4H,F] bf16
    const u16* __restrict__ Whh,        // [4H,H] bf16
    const float* __restrict__ bsum,     // [4H]
    float* __restrict__ Cst,            // [B,H] cell state
    int t)
{
    __shared__ __align__(16) u16 As[BATCH][APAD];       // 193 KB staged A
    __shared__ float gl2[2][4][BATCH][16];              // 32 KB partial gates

    const int tid  = threadIdx.x;
    const int lane = tid & 31;
    const int w    = tid >> 5;
    const int ln   = lane & 15;
    const int hi   = lane >> 4;
    const int kl   = hi * 8;
    const int ko   = hi * 16;
    const int g    = w & 3;
    const int kh   = w >> 2;
    const int blk  = blockIdx.x;

    // ---- cooperative stage: As[m][0..511]=x_t row m, As[m][512..1535]=h_{t-1} row m
    {
        const uint4 z = {0u, 0u, 0u, 0u};
        #pragma unroll 8
        for (int i = 0; i < 48; ++i) {
            int q  = i * 256 + tid;          // 0..12287 uint4 slots
            int m  = q / 192;                // 192 uint4 per row
            int c8 = q - m * 192;
            uint4 v;
            if (c8 < 64) {
                v = *(const uint4*)(Xbf + ((size_t)t * BATCH + m) * FDIM + c8 * 8);
            } else if (t > 0) {
                v = *(const uint4*)(Hs + ((size_t)(t - 1) * BATCH + m) * HDIM + (c8 - 64) * 8);
            } else {
                v = z;                        // h_{-1} == 0
            }
            *(uint4*)(&As[m][c8 * 8]) = v;
        }
    }
    __syncthreads();

    const int n = g * HDIM + blk * 16 + ln;          // weight row (gate column)
    const u16* wihrow = Wih + (size_t)n * FDIM;
    const u16* whhrow = Whh + (size_t)n * HDIM;

    const u16* arow[4];
    #pragma unroll
    for (int mt = 0; mt < 4; ++mt) arow[mt] = &As[mt * 16 + ln][0];

    const v8f zero = {0.f,0.f,0.f,0.f,0.f,0.f,0.f,0.f};
    v8f acc[4];
    #pragma unroll
    for (int mt = 0; mt < 4; ++mt) acc[mt] = zero;

    if (kh == 0) {
        for (int k0 = 0; k0 < 512; k0 += 32) {               // x part (W_ih)
            v16bf b = frag_b(wihrow, k0, ko);
            #pragma unroll
            for (int mt = 0; mt < 4; ++mt)
                acc[mt] = wmma_bf16(frag_a(arow[mt], k0, kl), b, acc[mt]);
        }
        for (int k0 = 512; k0 < 768; k0 += 32) {             // h part, K 0..255
            v16bf b = frag_b(whhrow, k0 - 512, ko);
            #pragma unroll
            for (int mt = 0; mt < 4; ++mt)
                acc[mt] = wmma_bf16(frag_a(arow[mt], k0, kl), b, acc[mt]);
        }
    } else {
        for (int k0 = 768; k0 < 1536; k0 += 32) {            // h part, K 256..1023
            v16bf b = frag_b(whhrow, k0 - 512, ko);
            #pragma unroll
            for (int mt = 0; mt < 4; ++mt)
                acc[mt] = wmma_bf16(frag_a(arow[mt], k0, kl), b, acc[mt]);
        }
    }

    // D layout: lane col ln, VGPR r -> row r + hi*8
    #pragma unroll
    for (int mt = 0; mt < 4; ++mt)
        #pragma unroll
        for (int r = 0; r < 8; ++r)
            gl2[kh][g][mt * 16 + r + kl][ln] = acc[mt][r];
    __syncthreads();

    // ---- pointwise cell update: 1024 outputs, 4 per thread
    #pragma unroll
    for (int j = 0; j < 4; ++j) {
        int e   = j * 256 + tid;
        int m   = e >> 4;
        int ncl = e & 15;
        int nh  = blk * 16 + ncl;
        float ig = gl2[0][0][m][ncl] + gl2[1][0][m][ncl] + bsum[nh];
        float fg = gl2[0][1][m][ncl] + gl2[1][1][m][ncl] + bsum[HDIM + nh];
        float gg = gl2[0][2][m][ncl] + gl2[1][2][m][ncl] + bsum[2 * HDIM + nh];
        float og = gl2[0][3][m][ncl] + gl2[1][3][m][ncl] + bsum[3 * HDIM + nh];
        float cprev = Cst[m * HDIM + nh];
        float cn = sigmoidf_(fg) * cprev + sigmoidf_(ig) * tanhf(gg);
        float hn = sigmoidf_(og) * tanhf(cn);
        Cst[m * HDIM + nh] = cn;
        Hs[((size_t)t * BATCH + m) * HDIM + nh] = f32_to_bf16(hn);
    }
}

// ---------- fused MLP head over all timesteps ----------
// grid = 512 WGs (one timestep = 64 rows each), block = 256 (8 waves).
// Hs[t] block (128 KB) staged once in LDS -> removes 8x A-fragment redundancy.
__global__ __launch_bounds__(256) void k_head(
    const u16* __restrict__ Hs,     // [T,B,H] bf16
    const u16* __restrict__ W1b,    // [128,H] bf16
    const float* __restrict__ b1,   // [128]
    const u16* __restrict__ W2b,    // [F,128] bf16
    const float* __restrict__ b2,   // [F]
    float* __restrict__ out)        // [T,B,F]
{
    __shared__ __align__(16) u16 Hst[BATCH][HPAD];     // 129 KB staged h block
    __shared__ __align__(16) u16 h1[BATCH][H1PAD];     // 17 KB selu(h@W1^T+b1)

    const int t    = blockIdx.x;
    const int tid  = threadIdx.x;
    const int lane = tid & 31;
    const int w    = tid >> 5;
    const int ln   = lane & 15;
    const int hi   = lane >> 4;
    const int kl   = hi * 8;
    const int ko   = hi * 16;

    // ---- cooperative stage of Hs[t] (64 x 1024 bf16)
    #pragma unroll 8
    for (int i = 0; i < 32; ++i) {
        int q  = i * 256 + tid;              // 0..8191 uint4 slots (128 per row)
        int m  = q >> 7;
        int c8 = q & 127;
        *(uint4*)(&Hst[m][c8 * 8]) =
            *(const uint4*)(Hs + ((size_t)t * BATCH + m) * HDIM + c8 * 8);
    }
    __syncthreads();

    const v8f zero = {0.f,0.f,0.f,0.f,0.f,0.f,0.f,0.f};

    // phase 1: wave w computes h1 columns [w*16, w*16+16), all 4 m-tiles, K=1024
    const int n1 = w * 16 + ln;
    const u16* w1row = W1b + (size_t)n1 * HDIM;
    const u16* hrow[4];
    #pragma unroll
    for (int mt = 0; mt < 4; ++mt) hrow[mt] = &Hst[mt * 16 + ln][0];

    v8f acc[4];
    #pragma unroll
    for (int mt = 0; mt < 4; ++mt) acc[mt] = zero;

    for (int k0 = 0; k0 < HDIM; k0 += 32) {
        v16bf b = frag_b(w1row, k0, ko);
        #pragma unroll
        for (int mt = 0; mt < 4; ++mt)
            acc[mt] = wmma_bf16(frag_a(hrow[mt], k0, kl), b, acc[mt]);
    }
    const float bias1 = b1[n1];
    #pragma unroll
    for (int mt = 0; mt < 4; ++mt) {
        #pragma unroll
        for (int r = 0; r < 8; ++r) {
            float v = acc[mt][r] + bias1;
            float s = (v > 0.f) ? 1.0507009873554805f * v
                                : 1.7580993408473766f * (__expf(v) - 1.f); // scale*alpha
            h1[mt * 16 + r + kl][n1] = f32_to_bf16(s);
        }
    }
    __syncthreads();

    // phase 2: wave w computes out columns [w*64, w*64+64), K=128 from LDS
    const u16* arow[4];
    #pragma unroll
    for (int mt = 0; mt < 4; ++mt) arow[mt] = &h1[mt * 16 + ln][0];

    for (int ntl = 0; ntl < 4; ++ntl) {
        const int n2 = (w * 4 + ntl) * 16 + ln;        // 0..511
        const u16* w2row = W2b + (size_t)n2 * MLPDIM;
        v8f oacc[4];
        #pragma unroll
        for (int mt = 0; mt < 4; ++mt) oacc[mt] = zero;

        #pragma unroll
        for (int k0 = 0; k0 < MLPDIM; k0 += 32) {
            v16bf b = frag_b(w2row, k0, ko);
            #pragma unroll
            for (int mt = 0; mt < 4; ++mt)
                oacc[mt] = wmma_bf16(frag_a(arow[mt], k0, kl), b, oacc[mt]);
        }
        const float bias2 = b2[n2];
        #pragma unroll
        for (int mt = 0; mt < 4; ++mt) {
            #pragma unroll
            for (int r = 0; r < 8; ++r) {
                int m = mt * 16 + r + kl;
                out[((size_t)t * BATCH + m) * FDIM + n2] = oacc[mt][r] + bias2;
            }
        }
    }
}

// ---------- host side ----------

extern "C" void kernel_launch(void* const* d_in, const int* in_sizes, int n_in,
                              void* d_out, int out_size, void* d_ws, size_t ws_size,
                              hipStream_t stream) {
    const float* x    = (const float*)d_in[0];
    const float* W_ih = (const float*)d_in[1];
    const float* W_hh = (const float*)d_in[2];
    const float* b_ih = (const float*)d_in[3];
    const float* b_hh = (const float*)d_in[4];
    const float* W1   = (const float*)d_in[5];
    const float* b1   = (const float*)d_in[6];
    const float* W2   = (const float*)d_in[7];
    const float* b2   = (const float*)d_in[8];
    float* out = (float*)d_out;
    char* ws = (char*)d_ws;

    // workspace layout (bytes, all 256-aligned)
    size_t off = 0;
    u16* Xbf   = (u16*)(ws + off); off += (size_t)T_STEPS * BATCH * FDIM * 2;     // 32 MB
    u16* Hs    = (u16*)(ws + off); off += (size_t)T_STEPS * BATCH * HDIM * 2;     // 64 MB
    u16* Wihb  = (u16*)(ws + off); off += (size_t)4 * HDIM * FDIM * 2;            // 4 MB
    u16* Whhb  = (u16*)(ws + off); off += (size_t)4 * HDIM * HDIM * 2;            // 8 MB
    u16* W1b   = (u16*)(ws + off); off += (size_t)MLPDIM * HDIM * 2;
    u16* W2b   = (u16*)(ws + off); off += (size_t)FDIM * MLPDIM * 2;
    float* bsum = (float*)(ws + off); off += (size_t)4 * HDIM * 4;
    float* Cst  = (float*)(ws + off); off += (size_t)BATCH * HDIM * 4;

    // one-time conversions (bf16 weights stay resident in 192MB L2 across all steps)
    k_cvt_bf16<<<8192, 256, 0, stream>>>(x,    Xbf,  T_STEPS * BATCH * FDIM);
    k_cvt_bf16<<<2048, 256, 0, stream>>>(W_ih, Wihb, 4 * HDIM * FDIM);
    k_cvt_bf16<<<2048, 256, 0, stream>>>(W_hh, Whhb, 4 * HDIM * HDIM);
    k_cvt_bf16<<<128,  256, 0, stream>>>(W1,   W1b,  MLPDIM * HDIM);
    k_cvt_bf16<<<64,   256, 0, stream>>>(W2,   W2b,  FDIM * MLPDIM);
    k_bias_and_zero<<<256, 256, 0, stream>>>(b_ih, b_hh, bsum, Cst);

    // sequential recurrence: 512 chained fused step kernels
    for (int t = 0; t < T_STEPS; ++t)
        k_lstm_step<<<64, 256, 0, stream>>>(Xbf, Hs, Wihb, Whhb, bsum, Cst, t);

    // deferred batched head over all T*B rows
    k_head<<<T_STEPS, 256, 0, stream>>>(Hs, W1b, b1, W2b, b2, out);
}
